// FeedForwardQuantum_65481071397465
// MI455X (gfx1250) — matrix operand order repaired and checked by reading, not compiled, not run on hardware
//
#include <hip/hip_runtime.h>

typedef __attribute__((ext_vector_type(16))) _Float16 v16h;
typedef __attribute__((ext_vector_type(8)))  float    v8f;

#define EMBED 256
#define FFN   1024
#define TOKENS (16 * 8192)

// ---------------------------------------------------------------------------
// Kernel A: Wc = W1 @ W2 (256x256), written in WMMA-B-fragment order (f16),
//           bc = b1 @ W2 + b2 (f32).
// Fragment order: for K-chunk c (32 K values) and N-tile nt (16 cols),
//   lane l holds column N = nt*16 + (l&15), halves j=0..15 = K = 32c + j + 16*(l>=16)
//   flat half index = ((nt*8 + c)*32 + l)*16 + j
// ---------------------------------------------------------------------------
__global__ __launch_bounds__(256) void combine_weights(
    const float* __restrict__ W1, const float* __restrict__ b1,
    const float* __restrict__ W2, const float* __restrict__ b2,
    _Float16* __restrict__ wcPacked, float* __restrict__ bc)
{
    __shared__ float row[FFN];
    const int n = threadIdx.x;     // output column 0..255
    const int k = blockIdx.x;      // input row 0..255 (block 256 = bias)

    if (k < EMBED) {
        for (int i = n; i < FFN; i += 256) row[i] = W1[k * FFN + i];
        __syncthreads();
        float acc = 0.0f;
#pragma unroll 4
        for (int f = 0; f < FFN; ++f) acc += row[f] * W2[f * EMBED + n];
        const int kt = k >> 5, kk = k & 31;
        const int nt = n >> 4, nn = n & 15;
        const int l  = nn + ((kk >= 16) ? 16 : 0);
        const int j  = kk & 15;
        wcPacked[(((nt * 8 + kt) * 32 + l) * 16) + j] = (_Float16)acc;
    } else {
        for (int i = n; i < FFN; i += 256) row[i] = b1[i];
        __syncthreads();
        float acc = 0.0f;
#pragma unroll 4
        for (int f = 0; f < FFN; ++f) acc += row[f] * W2[f * EMBED + n];
        bc[n] = acc + b2[n];
    }
}

// ---------------------------------------------------------------------------
// Kernel B: fused  relu(cumprod(cos(x))) @ Wc + bc
// 128 threads = 4 waves; each wave owns 32 tokens = two M=16 WMMA tiles.
// Phase 1: lane-per-token serial scan -> LDS, stored in A-fragment order.
//   16-bit A 16x32 order here: lane<16 : halves 0-15 -> K 0-15
//                          lane>=16: halves 0-15 -> K 16-31
// Phase 2: per n0 tile, accumulate 8 K-chunks with v_wmma_f32_16x16x32_f16.
// ---------------------------------------------------------------------------
__global__ __launch_bounds__(128) void ffn_fused(
    const float* __restrict__ x, const _Float16* __restrict__ wc,
    const float* __restrict__ bc, float* __restrict__ out)
{
    // [wave][chunk][tile*32 + consumer_lane][half]  -> 4*8*64*16*2B = 64 KB
    __shared__ __align__(32) _Float16 hbuf[4][8][64][16];

    const int tid  = threadIdx.x;
    const int w    = tid >> 5;
    const int lane = tid & 31;

    const int tokBase = blockIdx.x * 128 + w * 32;   // wave's first token
    const float* xp = x + (size_t)(tokBase + lane) * EMBED;

    // ---- Phase 1: scan (lane = token-in-wave) ----
    {
        const int t = lane >> 4;      // which M-tile this token belongs to
        const int m = lane & 15;      // row within the tile
        unsigned int* hb = (unsigned int*)&hbuf[w][0][0][0];
        float p = 1.0f;
#pragma unroll 1
        for (int c = 0; c < 8; ++c) {
#pragma unroll
            for (int i = 0; i < 32; i += 2) {
                const float x0 = xp[c * 32 + i];
                const float x1 = xp[c * 32 + i + 1];
                p *= __cosf(x0); const float h0 = fmaxf(p, 0.0f);
                p *= __cosf(x1); const float h1 = fmaxf(p, 0.0f);
                union { _Float16 h[2]; unsigned int u; } pk;
                pk.h[0] = (_Float16)h0;
                pk.h[1] = (_Float16)h1;
                // A-fragment scatter (i is compile-time constant here)
                const int g  = i >> 3;                     // group of 8 K values
                const int cl = m + ((g >> 1) << 4);        // consumer lane
                const int j  = (i & 7) + ((g & 1) << 3);   // half index (even)
                hb[(c * 64 + t * 32 + cl) * 8 + (j >> 1)] = pk.u;
            }
        }
    }
    __syncthreads();

    // ---- Phase 2: WMMA accumulate over K, tile N in 16 chunks ----
    const v16h* wcv = (const v16h*)wc;
    float* outBase = out + (size_t)tokBase * EMBED;
    const int colInTile = lane & 15;
    const int rowOff    = (lane >= 16) ? 8 : 0;

#pragma unroll 1
    for (int n0 = 0; n0 < 16; ++n0) {
        v8f acc0 = {0.f, 0.f, 0.f, 0.f, 0.f, 0.f, 0.f, 0.f};
        v8f acc1 = {0.f, 0.f, 0.f, 0.f, 0.f, 0.f, 0.f, 0.f};
#pragma unroll
        for (int c = 0; c < 8; ++c) {
            const v16h bfrag = wcv[(n0 * 8 + c) * 32 + lane];          // L2-resident
            const v16h af0 = *(const v16h*)&hbuf[w][c][lane][0];        // tile 0
            const v16h af1 = *(const v16h*)&hbuf[w][c][32 + lane][0];   // tile 1
            acc0 = __builtin_amdgcn_wmma_f32_16x16x32_f16(
                       false, af0, false, bfrag, (short)0, acc0, false, false);
            acc1 = __builtin_amdgcn_wmma_f32_16x16x32_f16(
                       false, af1, false, bfrag, (short)0, acc1, false, false);
        }
        const float bias = bc[n0 * 16 + colInTile];
        const int col = n0 * 16 + colInTile;
#pragma unroll
        for (int r = 0; r < 8; ++r) {
            // C/D layout: VGPR r -> row r (+8 for lanes>=16); lane -> column
            outBase[(size_t)(r + rowOff) * EMBED + col]      = acc0[r] + bias;
            outBase[(size_t)(16 + r + rowOff) * EMBED + col] = acc1[r] + bias;
        }
    }
}

// ---------------------------------------------------------------------------
extern "C" void kernel_launch(void* const* d_in, const int* in_sizes, int n_in,
                              void* d_out, int out_size, void* d_ws, size_t ws_size,
                              hipStream_t stream) {
    const float* x  = (const float*)d_in[0];
    const float* W1 = (const float*)d_in[1];
    const float* b1 = (const float*)d_in[2];
    const float* W2 = (const float*)d_in[3];
    const float* b2 = (const float*)d_in[4];
    float* out = (float*)d_out;

    _Float16* wcPacked = (_Float16*)d_ws;                    // 256*256*2 = 128 KB
    float*    bcPtr    = (float*)((char*)d_ws + EMBED * EMBED * sizeof(_Float16));

    combine_weights<<<EMBED + 1, 256, 0, stream>>>(W1, b1, W2, b2, wcPacked, bcPtr);
    ffn_fused<<<TOKENS / 128, 128, 0, stream>>>(x, wcPacked, bcPtr, out);
}
